// CrossReferenceAttention_22325240004732
// MI455X (gfx1250) — matrix operand order repaired and checked by reading, not compiled, not run on hardware
//
#include <hip/hip_runtime.h>

// ---------------------------------------------------------------------------
// CrossReferenceAttention for MI455X (gfx1250, wave32, WMMA)
//   K1: score gather + softmax           -> attn (bf16, row-major)
//   K2a: hidden f32 -> bf16, pre-swizzled into WMMA B-fragment tiles
//   K2b: out_w  f32 -> bf16 (row-major; its B-frags are naturally contiguous)
//   K3: ctx  = attn @ hidden   (v_wmma_f32_16x16x32_bf16, b128-fed)
//   K4: proj = ctx @ out_w^T   (v_wmma_f32_16x16x32_bf16, b128-fed)
//   K5: out  = LayerNorm(hidden + proj + out_b)
// ---------------------------------------------------------------------------

typedef __attribute__((ext_vector_type(16))) __bf16 v16bf;
typedef __attribute__((ext_vector_type(8)))  float  v8f;

#define NB   8          // batch
#define NS   1024       // sequence
#define NH   768        // hidden
#define LDV  8193       // cross_ref_matrix leading dim (V+1)

// 16 bf16 = 8 dwords = 2 x b128 = one WMMA A/B fragment
union FragBf {
  uint4    q[2];
  unsigned u[8];
  v16bf    v;
};

__device__ __forceinline__ unsigned short bf16_bits(float f) {
  union { __bf16 h; unsigned short s; } u;
  u.h = (__bf16)f;            // native v_cvt (RNE)
  return u.s;
}
__device__ __forceinline__ unsigned bf16_pack2(float lo, float hi) {
  return (unsigned)bf16_bits(lo) | ((unsigned)bf16_bits(hi) << 16);
}

// A-fragment (16x32 bf16, row-major source): two contiguous b128 loads.
// Contiguous K order (matches B): VGPR v -> K = 16*half + 2v
__device__ __forceinline__ void load_afrag(const unsigned short* __restrict__ row,
                                           int k0, int half, FragBf& f) {
  f.q[0] = *(const uint4*)(row + k0 + 16 * half);
  f.q[1] = *(const uint4*)(row + k0 + 16 * half + 8);
}

// ---------------------------------------------------------------------------
// K1: one block per (b, i) row. Gather scores, masked softmax, emit bf16 row.
// ---------------------------------------------------------------------------
__global__ void verse_softmax_kernel(const float* __restrict__ crm,
                                     const int*   __restrict__ ids,
                                     unsigned short* __restrict__ attn) {
  const int b   = blockIdx.y;
  const int i   = blockIdx.x;
  const int tid = threadIdx.x;              // 256 threads = 8 waves

  __shared__ int   s_ids[NS];
  __shared__ float s_row[NS];
  __shared__ float s_red[256];

  for (int j = tid; j < NS; j += 256) s_ids[j] = ids[b * NS + j];
  __syncthreads();

  const int  ri    = s_ids[i];
  const bool has_i = ri > 0;
  const float scale = 0.03608439182435161f;  // 1/sqrt(768)

  float lmax = -3.4e38f;
  for (int j = tid; j < NS; j += 256) {
    float v = 0.0f;
    if (has_i) {
      const int cj = s_ids[j];
      if (cj > 0) v = crm[(size_t)ri * LDV + cj] * scale;
    }
    s_row[j] = v;
    lmax = fmaxf(lmax, v);
  }
  s_red[tid] = lmax;
  __syncthreads();
  for (int off = 128; off > 0; off >>= 1) {
    if (tid < off) s_red[tid] = fmaxf(s_red[tid], s_red[tid + off]);
    __syncthreads();
  }
  const float m = s_red[0];
  __syncthreads();

  float lsum = 0.0f;
  for (int j = tid; j < NS; j += 256) {
    const float e = __expf(s_row[j] - m);
    s_row[j] = e;
    lsum += e;
  }
  s_red[tid] = lsum;
  __syncthreads();
  for (int off = 128; off > 0; off >>= 1) {
    if (tid < off) s_red[tid] += s_red[tid + off];
    __syncthreads();
  }
  const float inv = 1.0f / s_red[0];

  unsigned short* arow = attn + ((size_t)(b * NS + i)) * NS;
  for (int j = tid; j < NS; j += 256) arow[j] = bf16_bits(s_row[j] * inv);
}

// ---------------------------------------------------------------------------
// K2a: hidden f32 -> bf16, swizzled into B-fragment tiles.
// Tiled dword index: (((kb*NH + n) * 2 + half) * 8 + v) holds
// K = 32*kb + 16*half + 2v (,+1) at column n.
// One thread per (b, kb, n): 32 coalesced f32 reads, 4 x b128 writes.
// ---------------------------------------------------------------------------
__global__ void cvt_hidden_tiled_kernel(const float* __restrict__ in,
                                        unsigned* __restrict__ out) {
  const int tid = threadIdx.x;
  const int n   = blockIdx.x * 256 + tid;   // 0..767
  const int kb  = blockIdx.y;               // 0..31
  const int b   = blockIdx.z;

  const float* src = in + ((size_t)b * NS + (size_t)kb * 32) * NH + n;

  unsigned d[16];
  #pragma unroll
  for (int h = 0; h < 2; ++h) {
    #pragma unroll
    for (int v = 0; v < 8; ++v) {
      const int k = 16 * h + 2 * v;
      d[h * 8 + v] = bf16_pack2(src[(size_t)k * NH], src[(size_t)(k + 1) * NH]);
    }
  }

  unsigned* dst = out + (size_t)b * (NS / 32) * NH * 16 +
                  ((size_t)kb * NH + n) * 16;
  #pragma unroll
  for (int t = 0; t < 4; ++t)
    ((uint4*)dst)[t] = ((const uint4*)d)[t];
}

// ---------------------------------------------------------------------------
// K2b: elementwise f32 -> bf16 (out_w stays row-major)
// ---------------------------------------------------------------------------
__global__ void cvt_bf16_kernel(const float* __restrict__ in,
                                unsigned short* __restrict__ out, int n) {
  const int i = blockIdx.x * blockDim.x + threadIdx.x;
  if (i < n) out[i] = bf16_bits(in[i]);
}

// ---------------------------------------------------------------------------
// K3: per-batch ctx = attn(1024x1024 bf16) @ hidden(tiled bf16), f32 accum.
// Block = 8 waves; each wave owns a 16x16 tile; block covers 16(M) x 128(N).
// All fragment traffic is b128.
// ---------------------------------------------------------------------------
__global__ void attn_ctx_gemm_kernel(const unsigned short* __restrict__ attn,
                                     const unsigned* __restrict__ hidT,
                                     unsigned short* __restrict__ ctx) {
  const int lane   = threadIdx.x & 31;
  const int wave   = threadIdx.x >> 5;
  const int half   = lane >> 4;
  const int lane16 = lane & 15;

  const int b  = blockIdx.z;
  const int mt = blockIdx.y * 16;
  const int nt = blockIdx.x * 128 + wave * 16;
  const int n  = nt + lane16;

  const unsigned short* Arow = attn + (size_t)b * NS * NS + (size_t)(mt + lane16) * NS;
  const unsigned* Bt = hidT + (size_t)b * (NS / 32) * NH * 16;

  v8f acc = {};
  for (int kb = 0; kb < NS / 32; ++kb) {
    FragBf af, bf;
    load_afrag(Arow, kb * 32, half, af);
    const unsigned* bp = Bt + (((size_t)kb * NH + n) * 2 + half) * 8;
    bf.q[0] = ((const uint4*)bp)[0];
    bf.q[1] = ((const uint4*)bp)[1];
    acc = __builtin_amdgcn_wmma_f32_16x16x32_bf16(false, af.v, false, bf.v,
                                                  (short)0, acc, false, false);
  }

  unsigned short* C = ctx + (size_t)b * NS * NH;
  #pragma unroll
  for (int r = 0; r < 8; ++r) {
    const int m = mt + 8 * half + r;          // D layout: VGPR r -> M = r + 8*half
    C[(size_t)m * NH + n] = bf16_bits(acc[r]);
  }
}

// ---------------------------------------------------------------------------
// K4: proj = ctx(8192x768 bf16) @ out_w^T. B[k][n] = w[n*768+k], so the B
// fragment is 32 contiguous bytes of row n of out_w -> two b128 loads.
// ---------------------------------------------------------------------------
__global__ void proj_gemm_kernel(const unsigned short* __restrict__ ctx,
                                 const unsigned short* __restrict__ w,
                                 float* __restrict__ proj) {
  const int lane   = threadIdx.x & 31;
  const int wave   = threadIdx.x >> 5;
  const int half   = lane >> 4;
  const int lane16 = lane & 15;

  const int mt = blockIdx.y * 16;                 // M = B*S = 8192
  const int nt = blockIdx.x * 128 + wave * 16;    // N = 768

  const unsigned short* Arow = ctx + (size_t)(mt + lane16) * NH;
  const unsigned short* Wrow = w   + (size_t)(nt + lane16) * NH;

  v8f acc = {};
  for (int k0 = 0; k0 < NH; k0 += 32) {
    FragBf af, bf;
    load_afrag(Arow, k0, half, af);
    const unsigned short* bp = Wrow + k0 + 16 * half;   // K = 16*half + 2v pairs
    bf.q[0] = ((const uint4*)bp)[0];
    bf.q[1] = ((const uint4*)bp)[1];
    acc = __builtin_amdgcn_wmma_f32_16x16x32_bf16(false, af.v, false, bf.v,
                                                  (short)0, acc, false, false);
  }

  #pragma unroll
  for (int r = 0; r < 8; ++r) {
    const int m = mt + 8 * half + r;
    proj[(size_t)m * NH + nt + lane16] = acc[r];
  }
}

// ---------------------------------------------------------------------------
// K5: out = LN(hidden + proj + out_b) * ln_w + ln_b  (one block per row)
// ---------------------------------------------------------------------------
__global__ void residual_ln_kernel(const float* __restrict__ hidden,
                                   const float* __restrict__ proj,
                                   const float* __restrict__ out_b,
                                   const float* __restrict__ ln_w,
                                   const float* __restrict__ ln_b,
                                   float* __restrict__ out) {
  const int row = blockIdx.x;       // 0 .. B*S-1
  const int tid = threadIdx.x;      // 256 threads, 3 elems each (768)

  __shared__ float s1[256];
  __shared__ float s2[256];

  const float* hp = hidden + (size_t)row * NH;
  const float* pp = proj   + (size_t)row * NH;

  float x[3];
  float sum = 0.0f, sq = 0.0f;
  #pragma unroll
  for (int t = 0; t < 3; ++t) {
    const int o = tid + t * 256;
    x[t] = hp[o] + pp[o] + out_b[o];
    sum += x[t];
    sq  += x[t] * x[t];
  }
  s1[tid] = sum; s2[tid] = sq;
  __syncthreads();
  for (int off = 128; off > 0; off >>= 1) {
    if (tid < off) { s1[tid] += s1[tid + off]; s2[tid] += s2[tid + off]; }
    __syncthreads();
  }
  const float mu   = s1[0] * (1.0f / NH);
  const float var  = s2[0] * (1.0f / NH) - mu * mu;
  const float rstd = rsqrtf(var + 1e-5f);

  float* op = out + (size_t)row * NH;
  #pragma unroll
  for (int t = 0; t < 3; ++t) {
    const int o = tid + t * 256;
    op[o] = (x[t] - mu) * rstd * ln_w[o] + ln_b[o];
  }
}

// ---------------------------------------------------------------------------
extern "C" void kernel_launch(void* const* d_in, const int* in_sizes, int n_in,
                              void* d_out, int out_size, void* d_ws, size_t ws_size,
                              hipStream_t stream) {
  const float* hidden = (const float*)d_in[0];   // (8,1024,768)
  const int*   ids    = (const int*)  d_in[1];   // (8,1024)
  const float* crm    = (const float*)d_in[2];   // (8193,8193)
  const float* out_w  = (const float*)d_in[3];   // (768,768)
  const float* out_b  = (const float*)d_in[4];   // (768,)
  const float* ln_w   = (const float*)d_in[5];
  const float* ln_b   = (const float*)d_in[6];

  // Workspace layout (bytes):
  //   [0          , 16777216) attn  bf16 row-major (B*S*S)   -- dead after K3
  //   [16777216   , 29360128) hidden bf16, WMMA-B tiled      -- dead after K3
  //   [29360128   , 41943040) ctx   bf16 row-major (B*S*H)
  //   [41943040   , 43122688) w     bf16 row-major (H*H)
  //   proj f32 (B*S*H*4 = 25165824) reuses [0, 29360128)
  char* ws = (char*)d_ws;
  unsigned short* attn_bf = (unsigned short*)(ws);
  unsigned*       hid_t   = (unsigned*)(ws + 16777216u);
  unsigned short* ctx_bf  = (unsigned short*)(ws + 29360128u);
  unsigned short* w_bf    = (unsigned short*)(ws + 41943040u);
  float*          proj    = (float*)(ws);

  // K1: gather + softmax -> attn bf16
  verse_softmax_kernel<<<dim3(NS, NB), 256, 0, stream>>>(crm, ids, attn_bf);

  // K2: converts
  cvt_hidden_tiled_kernel<<<dim3(NH / 256, NS / 32, NB), 256, 0, stream>>>(hidden, hid_t);
  cvt_bf16_kernel<<<(NH * NH + 255) / 256, 256, 0, stream>>>(out_w, w_bf, NH * NH);

  // K3: ctx = attn @ hidden   (per batch 1024x768x1024)
  attn_ctx_gemm_kernel<<<dim3(NH / 128, NS / 16, NB), 256, 0, stream>>>(attn_bf, hid_t, ctx_bf);

  // K4: proj = ctx @ out_w^T  (8192x768x768)
  proj_gemm_kernel<<<dim3(NH / 128, (NB * NS) / 16, 1), 256, 0, stream>>>(ctx_bf, w_bf, proj);

  // K5: residual + layernorm -> d_out (f32)
  residual_ln_kernel<<<NB * NS, 256, 0, stream>>>(hidden, proj, out_b, ln_w, ln_b, (float*)d_out);
}
